// MultiHead_19971597926912
// MI455X (gfx1250) — matrix operand order repaired and checked by reading, not compiled, not run on hardware
//
#include <hip/hip_runtime.h>
#include <hip/hip_bf16.h>

// ---------------------------------------------------------------------------
// MultiHead attention for MI455X (gfx1250), wave32 + WMMA bf16.
// B=2, S=2048, D=1024, H=16, HD=64.
// ---------------------------------------------------------------------------

#define Bc 2
#define Sc 2048
#define Dc 1024
#define Hc 16
#define HDc 64

typedef __attribute__((ext_vector_type(16))) __bf16 v16bf;
typedef __attribute__((ext_vector_type(8)))  float  v8f;
typedef __attribute__((ext_vector_type(4)))  unsigned int u32x4;

__device__ __forceinline__ v16bf ld_frag2(const __bf16* p0, const __bf16* p1) {
    union { v16bf v; u32x4 q[2]; } r;
    r.q[0] = *(const u32x4*)p0;
    r.q[1] = *(const u32x4*)p1;
    return r.v;
}

__device__ __forceinline__ v8f wmma_bf16(v16bf a, v16bf b, v8f c) {
    // D(f32 16x16) = A(bf16 16x32) * B(bf16 32x16) + C
    return __builtin_amdgcn_wmma_f32_16x16x32_bf16(
        false, a, false, b, (short)0, c, false, false);
}

// ---------------------------------------------------------------------------
// Stage 1: W[k][n] (f32) -> WT[n][k] (bf16), 32x32 LDS-tiled, coalesced both ways
// ---------------------------------------------------------------------------
__global__ __launch_bounds__(256) void transpose_to_bf16(const float* __restrict__ W,
                                                         __bf16* __restrict__ WT) {
    __shared__ float tile[32][33];           // +1 pad kills bank conflicts
    const int tx = threadIdx.x & 31;
    const int ty = threadIdx.x >> 5;         // 0..7
    const int k0 = blockIdx.x * 32;
    const int n0 = blockIdx.y * 32;
#pragma unroll
    for (int i = 0; i < 4; ++i) {
        int kk = ty + i * 8;
        tile[kk][tx] = W[(size_t)(k0 + kk) * Dc + n0 + tx];   // coalesced read
    }
    __syncthreads();
#pragma unroll
    for (int i = 0; i < 4; ++i) {
        int nn = ty + i * 8;
        WT[(size_t)(n0 + nn) * Dc + k0 + tx] = (__bf16)tile[tx][nn];  // coalesced write
    }
}

// ---------------------------------------------------------------------------
// Stage 2/4: GEMM  C[M x N] = A[M x K] * Bt[N x K]^T + bias
//   A_F32   : A is fp32 (convert to bf16 while staging in LDS), else bf16
//   OUT_MODE: 0 = fp32 row-major [M][N]
//             1 = bf16  [B,H,S,HD]   (head-split Q/K)
//             2 = bf16  [B,H,HD,S]   (head-split + transposed V)
// Block = 256 threads (8 waves), block tile 128x128, K-step 64 (2 WMMA k-subs).
// Wave (wm = w&3, wn = w>>2) owns a 32x64 tile = 2x4 WMMA accumulators.
// 16 WMMAs per barrier pair.
// ---------------------------------------------------------------------------
template <bool A_F32, int OUT_MODE>
__global__ __launch_bounds__(256) void gemm_bf16_wmma(const void* __restrict__ Aptr,
                                                      const __bf16* __restrict__ Bt,
                                                      const float* __restrict__ bias,
                                                      void* __restrict__ Cout,
                                                      int Mdim, int Ndim, int Kdim) {
    __shared__ __bf16 As[128 * 64];   // 16 KB
    __shared__ __bf16 Bs[128 * 64];   // 16 KB

    const int tid  = threadIdx.x;
    const int lane = tid & 31;
    const int wave = tid >> 5;
    const int wm   = wave & 3;   // 4 waves along M
    const int wn   = wave >> 2;  // 2 waves along N
    const int l15  = lane & 15;
    const int hi   = lane >> 4;  // 0 | 1 (lane half)
    const int m0   = blockIdx.y * 128;
    const int n0   = blockIdx.x * 128;

    v8f acc[2][4];
#pragma unroll
    for (int i = 0; i < 2; ++i)
#pragma unroll
        for (int j = 0; j < 4; ++j)
#pragma unroll
            for (int r = 0; r < 8; ++r) acc[i][j][r] = 0.0f;

    for (int k0 = 0; k0 < Kdim; k0 += 64) {
        // ---- stage A tile (128 rows x 64 k) into LDS ----
        if (A_F32) {
            const float* A = (const float*)Aptr;
#pragma unroll
            for (int i = 0; i < 4; ++i) {
                int c   = tid + i * 256;        // 1024 chunks of 8 floats
                int row = c >> 3;
                int off = (c & 7) * 8;
                const float* src = A + (size_t)(m0 + row) * Kdim + k0 + off;
                union { u32x4 u; float f[4]; } a0, a1;
                a0.u = *(const u32x4*)(src);
                a1.u = *(const u32x4*)(src + 4);
                __bf16* dst = &As[row * 64 + off];
#pragma unroll
                for (int j = 0; j < 4; ++j) dst[j]     = (__bf16)a0.f[j];
#pragma unroll
                for (int j = 0; j < 4; ++j) dst[4 + j] = (__bf16)a1.f[j];
            }
        } else {
            const __bf16* A = (const __bf16*)Aptr;
#pragma unroll
            for (int i = 0; i < 4; ++i) {
                int c   = tid + i * 256;        // 1024 chunks of 8 bf16 (16B)
                int row = c >> 3;
                int off = (c & 7) * 8;
                *(u32x4*)&As[row * 64 + off] =
                    *(const u32x4*)(A + (size_t)(m0 + row) * Kdim + k0 + off);
            }
        }
        // ---- stage Bt tile (128 n-rows x 64 k) into LDS ----
#pragma unroll
        for (int i = 0; i < 4; ++i) {
            int c   = tid + i * 256;
            int row = c >> 3;
            int off = (c & 7) * 8;
            *(u32x4*)&Bs[row * 64 + off] =
                *(const u32x4*)(Bt + (size_t)(n0 + row) * Kdim + k0 + off);
        }
        __syncthreads();

        // ---- 2 k-substeps x (2x4 WMMA) ----
#pragma unroll
        for (int ks = 0; ks < 2; ++ks) {
            v16bf afrag[2], bfrag[4];
#pragma unroll
            for (int i = 0; i < 2; ++i) {
                // A 16x32 bf16: lanes 0-15 K {0..7,16..23}; lanes 16-31 K {8..15,24..31}
                const __bf16* base = &As[(wm * 32 + i * 16 + l15) * 64 + ks * 32 + hi * 8];
                afrag[i] = ld_frag2(base, base + 16);
            }
#pragma unroll
            for (int j = 0; j < 4; ++j) {
                // B 32x16 bf16 from [N][K] tile: lanes 0-15 K {0..7,16..23}; lanes 16-31 K {8..15,24..31}
                const __bf16* base = &Bs[(wn * 64 + j * 16 + l15) * 64 + ks * 32 + hi * 8];
                bfrag[j] = ld_frag2(base, base + 16);
            }
#pragma unroll
            for (int i = 0; i < 2; ++i)
#pragma unroll
                for (int j = 0; j < 4; ++j)
                    acc[i][j] = wmma_bf16(afrag[i], bfrag[j], acc[i][j]);
        }
        __syncthreads();
    }

    // ---- store: C element r -> (M = r + 8*hi, N = l15) ----
#pragma unroll
    for (int i = 0; i < 2; ++i) {
#pragma unroll
        for (int j = 0; j < 4; ++j) {
            int gm = m0 + wm * 32 + i * 16 + hi * 8;
            int gn = n0 + wn * 64 + j * 16 + l15;
            float bv = bias[gn];
#pragma unroll
            for (int r = 0; r < 8; ++r) {
                float val = acc[i][j][r] + bv;
                int row = gm + r;
                if (OUT_MODE == 0) {
                    ((float*)Cout)[(size_t)row * Ndim + gn] = val;
                } else {
                    int b  = row >> 11;          // row = b*S + s
                    int s  = row & (Sc - 1);
                    int h  = gn >> 6;            // gn = h*HD + hd
                    int hd = gn & (HDc - 1);
                    if (OUT_MODE == 1) {         // [B,H,S,HD]
                        ((__bf16*)Cout)[(((size_t)(b * Hc + h) * Sc + s) * HDc) + hd] = (__bf16)val;
                    } else {                     // [B,H,HD,S]
                        ((__bf16*)Cout)[(((size_t)(b * Hc + h) * HDc + hd) * Sc) + s] = (__bf16)val;
                    }
                }
            }
        }
    }
}

// ---------------------------------------------------------------------------
// Stage 3: flash attention. One wave (32 threads) per 16 query rows.
// 64-key chunks: 16 WMMAs per softmax round (vs 64 ds_bpermute), halving the
// shuffle+barrier cost per key relative to 32-key chunks.
//   Q,K : bf16 [B,H,S,HD]   Vt : bf16 [B,H,HD,S]   mask : f32 [B,S]
//   Ctx : bf16 [B*S, D] with column = h*HD + hd
// ---------------------------------------------------------------------------
__global__ __launch_bounds__(32) void attn_wmma(const __bf16* __restrict__ Q,
                                                const __bf16* __restrict__ K,
                                                const __bf16* __restrict__ Vt,
                                                const float* __restrict__ mask,
                                                __bf16* __restrict__ Ctx) {
    const int lane = threadIdx.x;
    const int l15  = lane & 15;
    const int hi   = lane >> 4;
    const int idx  = blockIdx.x;       // 0..4095
    const int qt   = idx & 127;        // S/16 query tiles
    const int bh   = idx >> 7;         // 0..31
    const int b    = bh >> 4;
    const int h    = bh & 15;

    const __bf16* Qp = Q  + ((size_t)bh * Sc + qt * 16) * HDc;
    const __bf16* Kp = K  + (size_t)bh * Sc * HDc;
    const __bf16* Vp = Vt + (size_t)bh * HDc * Sc;
    const float*  mp = mask + (size_t)b * Sc;

    __shared__ __bf16 Pl[16 * 64];     // P tile bounce buffer (C-layout -> A-layout)

    // Q fragments for HD=64 (two K=32 steps)
    v16bf aq[2];
#pragma unroll
    for (int kk = 0; kk < 2; ++kk) {
        const __bf16* base = Qp + l15 * HDc + kk * 32 + hi * 8;
        aq[kk] = ld_frag2(base, base + 16);
    }

    float m_run[8], l_run[8];
    v8f accv[4];
#pragma unroll
    for (int r = 0; r < 8; ++r) { m_run[r] = -3.0e38f; l_run[r] = 0.0f; }
#pragma unroll
    for (int j = 0; j < 4; ++j)
#pragma unroll
        for (int r = 0; r < 8; ++r) accv[j][r] = 0.0f;

    // Vt row base per lane for each hd tile (contiguous K=keys)
    const __bf16* vrow[4];
#pragma unroll
    for (int j = 0; j < 4; ++j) vrow[j] = Vp + (size_t)(j * 16 + l15) * Sc + hi * 8;

    const float inv_scale = 0.125f;    // 1/sqrt(64)

    for (int kc = 0; kc < Sc; kc += 64) {
        // ---- scores: 4 tiles of 16 keys, each = 2 WMMAs over HD ----
        v8f sc[4];
#pragma unroll
        for (int t = 0; t < 4; ++t) {
            int key0 = kc + t * 16;
            v8f c;
#pragma unroll
            for (int r = 0; r < 8; ++r) c[r] = 0.0f;
#pragma unroll
            for (int kk = 0; kk < 2; ++kk) {
                const __bf16* base = Kp + (size_t)(key0 + l15) * HDc + kk * 32 + hi * 8;
                v16bf bk = ld_frag2(base, base + 16);
                c = wmma_bf16(aq[kk], bk, c);
            }
            sc[t] = c;
        }
        // scale + mask (N = key = lane&15 within tile, same for all 8 elems)
        float mk[4];
#pragma unroll
        for (int t = 0; t < 4; ++t) mk[t] = mp[kc + t * 16 + l15] * -1e9f;
#pragma unroll
        for (int t = 0; t < 4; ++t)
#pragma unroll
            for (int r = 0; r < 8; ++r) sc[t][r] = sc[t][r] * inv_scale + mk[t];

        // ---- online softmax (row reductions across 16-lane halves) ----
        float scale8[8];
#pragma unroll
        for (int r = 0; r < 8; ++r) {
            float v = fmaxf(fmaxf(sc[0][r], sc[1][r]), fmaxf(sc[2][r], sc[3][r]));
#pragma unroll
            for (int i = 1; i < 16; i <<= 1) v = fmaxf(v, __shfl_xor(v, i, 32));
            float mnew = fmaxf(m_run[r], v);
            float sold = __expf(m_run[r] - mnew);
            m_run[r] = mnew;
            float s = 0.0f;
#pragma unroll
            for (int t = 0; t < 4; ++t) {
                float p = __expf(sc[t][r] - mnew);
                sc[t][r] = p;
                s += p;
            }
#pragma unroll
            for (int i = 1; i < 16; i <<= 1) s += __shfl_xor(s, i, 32);
            l_run[r] = l_run[r] * sold + s;
            scale8[r] = sold;
        }
#pragma unroll
        for (int j = 0; j < 4; ++j)
#pragma unroll
            for (int r = 0; r < 8; ++r) accv[j][r] *= scale8[r];

        // ---- C-layout P -> LDS [16 rows][64 keys] -> A fragments ----
#pragma unroll
        for (int t = 0; t < 4; ++t)
#pragma unroll
            for (int r = 0; r < 8; ++r)
                Pl[(r + hi * 8) * 64 + t * 16 + l15] = (__bf16)sc[t][r];
        __syncthreads();
        v16bf ap[2];
#pragma unroll
        for (int kk = 0; kk < 2; ++kk) {
            const __bf16* pbase = &Pl[l15 * 64 + kk * 32 + hi * 8];
            ap[kk] = ld_frag2(pbase, pbase + 16);
        }
        __syncthreads();

        // ---- P * V : 4 hd tiles x 2 key-substeps = 8 WMMAs ----
#pragma unroll
        for (int j = 0; j < 4; ++j) {
#pragma unroll
            for (int kk = 0; kk < 2; ++kk) {
                const __bf16* vb = vrow[j] + kc + kk * 32;
                v16bf bv = ld_frag2(vb, vb + 16);
                accv[j] = wmma_bf16(ap[kk], bv, accv[j]);
            }
        }
    }

    // ---- finalize: ctx = acc / l, write bf16 [B*S][D] ----
#pragma unroll
    for (int j = 0; j < 4; ++j) {
#pragma unroll
        for (int r = 0; r < 8; ++r) {
            int s = qt * 16 + r + hi * 8;
            float val = accv[j][r] / l_run[r];
            Ctx[((size_t)(b * Sc + s)) * Dc + h * HDc + j * 16 + l15] = (__bf16)val;
        }
    }
}

// ---------------------------------------------------------------------------
// Launch
// ---------------------------------------------------------------------------
extern "C" void kernel_launch(void* const* d_in, const int* in_sizes, int n_in,
                              void* d_out, int out_size, void* d_ws, size_t ws_size,
                              hipStream_t stream) {
    const float* q    = (const float*)d_in[0];
    const float* k    = (const float*)d_in[1];
    const float* v    = (const float*)d_in[2];
    const float* mask = (const float*)d_in[3];
    const float* wq   = (const float*)d_in[4];
    const float* bq   = (const float*)d_in[5];
    const float* wk   = (const float*)d_in[6];
    const float* bk   = (const float*)d_in[7];
    const float* wv   = (const float*)d_in[8];
    const float* bv   = (const float*)d_in[9];
    const float* wo   = (const float*)d_in[10];
    const float* bo   = (const float*)d_in[11];

    // workspace layout (bytes): 4x WT (2MB) | Qh 8MB | Kh 8MB | Vt 8MB | Ctx 8MB
    char* ws = (char*)d_ws;
    const size_t MB = 1024ull * 1024ull;
    __bf16* wqT = (__bf16*)(ws + 0 * MB);
    __bf16* wkT = (__bf16*)(ws + 2 * MB);
    __bf16* wvT = (__bf16*)(ws + 4 * MB);
    __bf16* woT = (__bf16*)(ws + 6 * MB);
    __bf16* Qh  = (__bf16*)(ws + 8 * MB);
    __bf16* Kh  = (__bf16*)(ws + 16 * MB);
    __bf16* Vt  = (__bf16*)(ws + 24 * MB);
    __bf16* Ctx = (__bf16*)(ws + 32 * MB);

    const int Mrows = Bc * Sc;   // 4096

    // Stage 1: weight transpose + bf16 convert (WT[n][k]), coalesced tiled
    dim3 tgrid(Dc / 32, Dc / 32);        // (32, 32)
    transpose_to_bf16<<<tgrid, 256, 0, stream>>>(wq, wqT);
    transpose_to_bf16<<<tgrid, 256, 0, stream>>>(wk, wkT);
    transpose_to_bf16<<<tgrid, 256, 0, stream>>>(wv, wvT);
    transpose_to_bf16<<<tgrid, 256, 0, stream>>>(wo, woT);

    // Stage 2: QKV projections (A fp32 -> bf16 in-LDS), head-split outputs
    dim3 ggrid(Dc / 128, Mrows / 128);   // (8, 32)
    gemm_bf16_wmma<true, 1><<<ggrid, 256, 0, stream>>>(q, wqT, bq, Qh, Mrows, Dc, Dc);
    gemm_bf16_wmma<true, 1><<<ggrid, 256, 0, stream>>>(k, wkT, bk, Kh, Mrows, Dc, Dc);
    gemm_bf16_wmma<true, 2><<<ggrid, 256, 0, stream>>>(v, wvT, bv, Vt, Mrows, Dc, Dc);

    // Stage 3: flash attention, one wave per 16 query rows, 64-key chunks
    attn_wmma<<<Bc * Hc * (Sc / 16), 32, 0, stream>>>(Qh, Kh, Vt, mask, Ctx);

    // Stage 4: output projection -> fp32 d_out
    gemm_bf16_wmma<false, 0><<<ggrid, 256, 0, stream>>>(Ctx, woT, bo, d_out, Mrows, Dc, Dc);
}
